// TimesNetASU_2362232012867
// MI455X (gfx1250) — matrix operand order repaired and checked by reading, not compiled, not run on hardware
//
#include <hip/hip_runtime.h>
#include <hip/hip_bf16.h>
#include <math.h>

typedef __attribute__((ext_vector_type(16))) _Float16 v16h;
typedef __attribute__((ext_vector_type(8)))  float    v8f;

#define TT   64   // time steps
#define DD   64   // model dim
#define BN   64   // B*NA sequences
#define NF   33   // rfft bins
#define LMAX 128  // max folded length (row LMAX = zero pad row in LDS)

// ---------------------------------------------------------------------------
// Build WMMA-fragment-ordered combined 5x5 weights.
// K order: kk = (ky*5+kx)*Cin + ci   (tap-major so each 32-K step is one tap)
// A-fragment layout (ISA 7.12.2, 16-bit A 16x32):
//   lane = hf*16 + m ; element j = 2*vi+q ; kbA = vi<4 ? 2vi+8hf : 16+2(vi-4)+8hf
//   Wd[ ((mt*KSTEP + ks)*32 + lane)*16 + j ] = Wcomb[co=mt*16+m][K=ks*32+kbA+q]
// ---------------------------------------------------------------------------
__global__ void k_prep(const float* w1, const float* b1,
                       const float* w3, const float* b3,
                       const float* w5, const float* b5,
                       int Cout, int Cin, _Float16* Wd, float* Bd)
{
    int idx = blockIdx.x * blockDim.x + threadIdx.x;
    int total = Cout * Cin * 25;
    if (idx < total) {
        int KSTEP = (Cin * 25) / 32;
        int j    = idx & 15;
        int lane = (idx >> 4) & 31;
        int ks   = (idx >> 9) % KSTEP;
        int mt   = (idx >> 9) / KSTEP;
        int m  = lane & 15, hf = lane >> 4;
        int vi = j >> 1,    q  = j & 1;
        int kbA = (vi < 4) ? (2 * vi + 8 * hf) : (16 + 2 * (vi - 4) + 8 * hf);
        int K   = ks * 32 + kbA + q;
        int co  = mt * 16 + m;
        int r   = K / Cin;
        int ci  = K - r * Cin;
        int ky  = r / 5, kx = r - ky * 5;
        float v = w5[((co * Cin + ci) * 5 + ky) * 5 + kx];
        if (ky >= 1 && ky <= 3 && kx >= 1 && kx <= 3)
            v += w3[((co * Cin + ci) * 3 + (ky - 1)) * 3 + (kx - 1)];
        if (ky == 2 && kx == 2)
            v += w1[co * Cin + ci];
        Wd[idx] = (_Float16)(v * (1.0f / 3.0f));
    }
    if (idx < Cout)
        Bd[idx] = (b1[idx] + b3[idx] + b5[idx]) * (1.0f / 3.0f);
}

// ---------------------------------------------------------------------------
// Embedding: X[bn][t][d] = inputs[bn][t][:16] @ emb_w + emb_b
// ---------------------------------------------------------------------------
__global__ void k_embed(const float* __restrict__ in, const float* __restrict__ ew,
                        const float* __restrict__ eb, float* __restrict__ X)
{
    int i = blockIdx.x * blockDim.x + threadIdx.x;
    if (i >= BN * TT * DD) return;
    int d  = i & 63;
    int bt = i >> 6;
    float s = eb[d];
    const float* ip = in + bt * 16;
    #pragma unroll
    for (int f = 0; f < 16; ++f) s += ip[f] * ew[f * DD + d];
    X[i] = s;
}

// ---------------------------------------------------------------------------
// Amplitude spectrum: amp[bn][f] = mean_d |rfft(X[bn,:,d])[f]|
// ---------------------------------------------------------------------------
__global__ void k_spec(const float* __restrict__ X, float* __restrict__ amp)
{
    __shared__ float sA[NF];
    int bn = blockIdx.x, tid = threadIdx.x;
    if (tid < NF) sA[tid] = 0.0f;
    __syncthreads();
    for (int i = tid; i < NF * DD; i += blockDim.x) {
        int f = i / DD, d = i - f * DD;
        float w = 6.28318530717958647f * (float)f / (float)TT;
        float re = 0.0f, im = 0.0f;
        for (int t = 0; t < TT; ++t) {
            float xv = X[bn * (TT * DD) + t * DD + d];
            re += xv * cosf(w * (float)t);
            im += xv * sinf(w * (float)t);
        }
        atomicAdd(&sA[f], sqrtf(re * re + im * im));
    }
    __syncthreads();
    if (tid < NF) amp[bn * NF + tid] = sA[tid] * (1.0f / (float)DD);
}

// ---------------------------------------------------------------------------
// Plan: freq = mean_bn amp (freq[0]=0); top-2 -> {i0,i1,p0,p1}; softmax weights
// ---------------------------------------------------------------------------
__global__ void k_plan(const float* __restrict__ amp, int* __restrict__ plan,
                       float* __restrict__ wsm)
{
    __shared__ float fr[NF];
    __shared__ int   sidx[2];
    int tid = threadIdx.x;
    if (tid < NF) {
        float s = 0.0f;
        for (int bn = 0; bn < BN; ++bn) s += amp[bn * NF + tid];
        fr[tid] = (tid == 0) ? 0.0f : s * (1.0f / (float)BN);
    }
    __syncthreads();
    if (tid == 0) {
        float b0 = -1e30f, b1 = -1e30f; int i0 = 1, i1 = 2;
        for (int i = 0; i < NF; ++i) {
            float v = fr[i];
            if (v > b0)      { b1 = b0; i1 = i0; b0 = v; i0 = i; }
            else if (v > b1) { b1 = v;  i1 = i; }
        }
        if (i0 < 1) i0 = 1;
        if (i1 < 1) i1 = 1;
        sidx[0] = i0; sidx[1] = i1;
        plan[0] = i0; plan[1] = i1;
        plan[2] = TT / i0; plan[3] = TT / i1;
    }
    __syncthreads();
    if (tid < BN) {
        float e0 = amp[tid * NF + sidx[0]];
        float e1 = amp[tid * NF + sidx[1]];
        float mx = fmaxf(e0, e1);
        float w0 = expf(e0 - mx), w1 = expf(e1 - mx);
        float s = w0 + w1;
        wsm[tid * 2 + 0] = w0 / s;
        wsm[tid * 2 + 1] = w1 / s;
    }
}

// ---------------------------------------------------------------------------
// Implicit-GEMM 5x5 conv via WMMA f32_16x16x32_f16, tap-major K order.
// LDS image [t][ci] with an extra all-zero row at t==LMAX: out-of-image taps
// are redirected there (tSafe = vOK ? tA : LMAX) -> fully branchless B loads.
// grid = (bn=64, ntile=8), block = (COUT/16) waves, one wave per 16x16 tile.
// FIRST: in f32 X[bn][t][c], GELU, out f16 Yout[bn][n][c]   (64 -> 128)
// !FIRST: in f16 Yin[bn][n][c], out f32 Rout[bn][t][c]      (128 -> 64)
// ---------------------------------------------------------------------------
template<int CIN, int COUT, bool FIRST>
__global__ __launch_bounds__((COUT / 16) * 32)
void k_conv(const float* __restrict__ Xin, const _Float16* __restrict__ Yin,
            const _Float16* __restrict__ Wc, const float* __restrict__ Bc,
            const int* __restrict__ plan, int kSel,
            _Float16* __restrict__ Yout, float* __restrict__ Rout)
{
    constexpr int KSTEP = (CIN * 25) / 32;
    constexpr int CS    = CIN / 32;            // K steps per tap
    constexpr int NWAVE = COUT / 16;
    __shared__ _Float16 lds[(LMAX + 1) * CIN]; // [t][ci], row LMAX == zeros

    const int bn    = blockIdx.x;
    const int ntile = blockIdx.y;
    const int tid   = threadIdx.x;
    const int lane  = tid & 31;
    const int wave  = tid >> 5;

    const int p = plan[2 + kSel];
    const int L = ((TT + p - 1) / p) * p;      // 64 <= L < 128
    const int H = L / p;
    if (ntile * 16 >= L) return;               // uniform per block

    // ---- stage folded image into LDS as [t][ci]; rows >= valid and the
    //      dummy row LMAX are zero-filled
    if (FIRST) {
        for (int i = tid; i < (LMAX + 1) * CIN; i += NWAVE * 32) {
            int t = i >> 6, ci = i & (CIN - 1);
            lds[i] = (t < TT) ? (_Float16)Xin[bn * (TT * DD) + t * DD + ci]
                              : (_Float16)0.0f;
        }
    } else {
        const uint4* src = (const uint4*)(Yin + (size_t)bn * LMAX * CIN);
        uint4* dst = (uint4*)lds;
        const uint4 z = make_uint4(0, 0, 0, 0);
        for (int i = tid; i < (LMAX + 1) * CIN / 8; i += NWAVE * 32) {
            int t = i / (CIN / 8);
            dst[i] = (t < L) ? src[i] : z;
        }
    }
    __syncthreads();

    const int  m     = lane & 15;
    const int  hf    = lane >> 4;
    const int  n     = ntile * 16 + m;         // output spatial column
    const int  hh0   = n / p, ww0 = n - (n / p) * p;
    const bool inImg = (n < L);

    const v16h* Ap = ((const v16h*)Wc) + (size_t)wave * KSTEP * 32 + lane;

    v8f acc = {};
    #pragma unroll
    for (int r = 0; r < 25; ++r) {
        const int dy = r / 5 - 2, dx = (r % 5) - 2;
        const int hh = hh0 + dy,  ww = ww0 + dx;
        const bool vOK = inImg && ((unsigned)hh < (unsigned)H)
                               && ((unsigned)ww < (unsigned)p);
        const int tSafe = vOK ? (hh * p + ww) : LMAX;   // zero row if OOB
        const _Float16* bp = &lds[tSafe * CIN + 8 * hf];
        #pragma unroll
        for (int cs = 0; cs < CS; ++cs) {
            const int ks = r * CS + cs;
            v16h a = Ap[ks * 32];
            typedef __attribute__((ext_vector_type(8))) _Float16 v8h_;
            v8h_ b0 = *(const v8h_*)(bp + cs * 32);
            v8h_ b1 = *(const v8h_*)(bp + cs * 32 + 16);
            v16h b = __builtin_shufflevector(b0, b1, 0,1,2,3,4,5,6,7,8,9,10,11,12,13,14,15);
            acc = __builtin_amdgcn_wmma_f32_16x16x32_f16(
                      false, a, false, b, (short)0, acc, false, false);
        }
    }

    // ---- epilogue: C/D layout: lane col n = lane%16, row m = vi + 8*hf
    const int co0 = wave * 16 + 8 * hf;        // 8 consecutive output channels
    if (FIRST) {
        union { _Float16 h[8]; uint4 u; } pk;
        #pragma unroll
        for (int vi = 0; vi < 8; ++vi) {
            float x = acc[vi] + Bc[co0 + vi];
            float g = 0.5f * x * (1.0f + erff(x * 0.70710678118654752f));
            pk.h[vi] = (_Float16)g;
        }
        if (inImg)
            *(uint4*)(Yout + (size_t)bn * LMAX * COUT + n * COUT + co0) = pk.u;
    } else {
        if (n < TT) {
            float4* rp = (float4*)(Rout + bn * (TT * DD) + n * DD + co0);
            rp[0] = make_float4(acc[0] + Bc[co0 + 0], acc[1] + Bc[co0 + 1],
                                acc[2] + Bc[co0 + 2], acc[3] + Bc[co0 + 3]);
            rp[1] = make_float4(acc[4] + Bc[co0 + 4], acc[5] + Bc[co0 + 5],
                                acc[6] + Bc[co0 + 6], acc[7] + Bc[co0 + 7]);
        }
    }
}

// ---------------------------------------------------------------------------
// Weighted combine + residual: X += w0*res0 + w1*res1
// ---------------------------------------------------------------------------
__global__ void k_comb(float* __restrict__ X, const float* __restrict__ r0,
                       const float* __restrict__ r1, const float* __restrict__ wsm)
{
    int i = blockIdx.x * blockDim.x + threadIdx.x;
    if (i >= BN * TT * DD) return;
    int bn = i >> 12;
    X[i] = X[i] + wsm[bn * 2] * r0[i] + wsm[bn * 2 + 1] * r1[i];
}

// ---------------------------------------------------------------------------
// Head: time-mean -> LayerNorm -> proj -> BatchNorm(axis0) -> sigmoid -> mask
// ---------------------------------------------------------------------------
__global__ void k_head(const float* __restrict__ X,
                       const float* __restrict__ ln_g, const float* __restrict__ ln_b,
                       const float* __restrict__ proj_w, const float* __restrict__ proj_b,
                       const float* __restrict__ bn_g, const float* __restrict__ bn_b,
                       const unsigned char* __restrict__ mask, float* __restrict__ out)
{
    __shared__ float sS[BN];
    int tid = threadIdx.x;
    if (tid < BN) {
        int bn = tid;
        float mu = 0.0f;
        for (int d = 0; d < DD; ++d) {
            float md = 0.0f;
            for (int t = 0; t < TT; ++t) md += X[bn * (TT * DD) + t * DD + d];
            mu += md * (1.0f / (float)TT);
        }
        mu *= (1.0f / (float)DD);
        float var = 0.0f;
        for (int d = 0; d < DD; ++d) {
            float md = 0.0f;
            for (int t = 0; t < TT; ++t) md += X[bn * (TT * DD) + t * DD + d];
            md *= (1.0f / (float)TT);
            float c = md - mu;
            var += c * c;
        }
        var *= (1.0f / (float)DD);
        float inv = rsqrtf(var + 1e-5f);
        float s = proj_b[0];
        for (int d = 0; d < DD; ++d) {
            float md = 0.0f;
            for (int t = 0; t < TT; ++t) md += X[bn * (TT * DD) + t * DD + d];
            md *= (1.0f / (float)TT);
            float h = (md - mu) * inv * ln_g[d] + ln_b[d];
            s += h * proj_w[d];
        }
        sS[bn] = s;
    }
    __syncthreads();
    if (tid < 32) {
        int na = tid;
        float s0 = sS[na], s1 = sS[32 + na];
        float bm = 0.5f * (s0 + s1);
        float bv = 0.5f * ((s0 - bm) * (s0 - bm) + (s1 - bm) * (s1 - bm));
        float inv = rsqrtf(bv + 1e-5f);
        for (int b = 0; b < 2; ++b) {
            float sv = b ? s1 : s0;
            float v  = (sv - bm) * inv * bn_g[na] + bn_b[na];
            float sg = 1.0f / (1.0f + expf(-v));
            int i = b * 32 + na;
            out[i] = mask[i] ? -INFINITY : sg;
        }
    }
}

// ---------------------------------------------------------------------------
extern "C" void kernel_launch(void* const* d_in, const int* in_sizes, int n_in,
                              void* d_out, int out_size, void* d_ws, size_t ws_size,
                              hipStream_t stream)
{
    (void)in_sizes; (void)n_in; (void)out_size; (void)ws_size;

    const float*         inputs = (const float*)d_in[0];
    const unsigned char* maskp  = (const unsigned char*)d_in[1];
    const float* emb_w  = (const float*)d_in[2];
    const float* emb_b  = (const float*)d_in[3];
    const float* ln_g   = (const float*)d_in[4];
    const float* ln_b   = (const float*)d_in[5];
    const float* proj_w = (const float*)d_in[6];
    const float* proj_b = (const float*)d_in[7];
    const float* bn_g   = (const float*)d_in[8];
    const float* bn_b   = (const float*)d_in[9];
    auto blkp = [&](int layer, int conv, int item) -> const float* {
        return (const float*)d_in[10 + layer * 12 + conv * 6 + item];
    };

    char*  base = (char*)d_ws;
    size_t off  = 0;
    auto alloc = [&](size_t bytes) -> char* {
        char* r = base + off;
        off = (off + bytes + 255) & ~(size_t)255;
        return r;
    };
    float*    X    = (float*)alloc((size_t)BN * TT * DD * 4);
    _Float16* Wc[2][2]; float* Bc[2][2];
    for (int l = 0; l < 2; ++l) {
        Wc[l][0] = (_Float16*)alloc((size_t)128 * 64 * 25 * 2);
        Bc[l][0] = (float*)alloc(128 * 4);
        Wc[l][1] = (_Float16*)alloc((size_t)64 * 128 * 25 * 2);
        Bc[l][1] = (float*)alloc(64 * 4);
    }
    float*    amp  = (float*)alloc((size_t)BN * NF * 4);
    int*      plan = (int*)alloc(16 * 4);
    float*    wsm  = (float*)alloc((size_t)BN * 2 * 4);
    _Float16* y0   = (_Float16*)alloc((size_t)BN * LMAX * 128 * 2);
    float*    res0 = (float*)alloc((size_t)BN * TT * DD * 4);
    float*    res1 = (float*)alloc((size_t)BN * TT * DD * 4);

    for (int l = 0; l < 2; ++l) {
        k_prep<<<(128 * 64 * 25 + 255) / 256, 256, 0, stream>>>(
            blkp(l, 0, 0), blkp(l, 0, 1), blkp(l, 0, 2), blkp(l, 0, 3),
            blkp(l, 0, 4), blkp(l, 0, 5), 128, 64, Wc[l][0], Bc[l][0]);
        k_prep<<<(64 * 128 * 25 + 255) / 256, 256, 0, stream>>>(
            blkp(l, 1, 0), blkp(l, 1, 1), blkp(l, 1, 2), blkp(l, 1, 3),
            blkp(l, 1, 4), blkp(l, 1, 5), 64, 128, Wc[l][1], Bc[l][1]);
    }

    k_embed<<<(BN * TT * DD + 255) / 256, 256, 0, stream>>>(inputs, emb_w, emb_b, X);

    for (int l = 0; l < 2; ++l) {
        k_spec<<<BN, 256, 0, stream>>>(X, amp);
        k_plan<<<1, 64, 0, stream>>>(amp, plan, wsm);
        for (int kSel = 0; kSel < 2; ++kSel) {
            float* rk = kSel ? res1 : res0;
            dim3 g(BN, LMAX / 16);
            k_conv<64, 128, true ><<<g, 256, 0, stream>>>(
                X, nullptr, Wc[l][0], Bc[l][0], plan, kSel, y0, nullptr);
            k_conv<128, 64, false><<<g, 128, 0, stream>>>(
                nullptr, y0, Wc[l][1], Bc[l][1], plan, kSel, nullptr, rk);
        }
        k_comb<<<(BN * TT * DD + 255) / 256, 256, 0, stream>>>(X, res0, res1, wsm);
    }

    k_head<<<1, 64, 0, stream>>>(X, ln_g, ln_b, proj_w, proj_b,
                                 bn_g, bn_b, maskp, (float*)d_out);
}
